// LiteMLA_13065290514879
// MI455X (gfx1250) — matrix operand order repaired and checked by reading, not compiled, not run on hardware
//
#include <hip/hip_runtime.h>
#include <hip/hip_bf16.h>

typedef __attribute__((ext_vector_type(16))) __bf16 v16bf;
typedef __attribute__((ext_vector_type(8)))  float  v8f;

// Problem constants (from reference)
#define BATCH 8
#define CIN   256
#define HWPX  4096      // 64*64
#define TD3   768       // 3*td
#define MSCH  1536      // concat(qkv, agg)
#define ATTCH 512       // 64 heads * 8

static __device__ inline unsigned pack_bf16x2(float lo, float hi) {
    union { __bf16 h; unsigned short u; } a, b;
    a.h = (__bf16)lo; b.h = (__bf16)hi;
    return ((unsigned)b.u << 16) | (unsigned)a.u;
}

// CDNA5 async global->LDS copy (ASYNCcnt-tracked), 16 bytes per lane.
static __device__ inline void async_copy_b128(unsigned lds_off, const void* gptr) {
    asm volatile("global_load_async_to_lds_b128 %0, %1, off"
                 :: "v"(lds_off), "v"(gptr) : "memory");
}
static __device__ inline void wait_async0() {
    asm volatile("s_wait_asynccnt 0x0" ::: "memory");
}

// ---------------------------------------------------------------------------
// Pack weight matrix A (MxK fp32, row-major) into WMMA A-fragment order, bf16:
// P[((mt*ksteps + ks)*32 + lane)*16 + e], e = fragment element (ISA 7.12.2):
//   lanes 0-15 : VGPR j -> K = {0..7,16..23};  lanes 16-31: K = {8..15,24..31}
// ---------------------------------------------------------------------------
__global__ __launch_bounds__(256) void pack_wgt(
    const float* __restrict__ A, __bf16* __restrict__ P, int M, int K)
{
    int tid  = blockIdx.x * 256 + threadIdx.x;
    int lane = tid & 31;
    int rest = tid >> 5;
    int ksteps = K >> 5;
    int ks = rest % ksteps;
    int mt = rest / ksteps;
    if (mt >= (M >> 4)) return;

    int half = lane >> 4, l = lane & 15;
    const float* arow = A + (size_t)(mt * 16 + l) * K;
    __bf16* out = P + ((size_t)(mt * ksteps + ks) * 32 + lane) * 16;
#pragma unroll
    for (int j = 0; j < 8; ++j) {
        int k0 = (ks << 5) + ((j & 4) << 2) + (half << 3) + ((j & 3) << 1);
        out[2 * j]     = (__bf16)arow[k0];
        out[2 * j + 1] = (__bf16)arow[k0 + 1];
    }
}

// ---------------------------------------------------------------------------
// Transpose + convert: in [b][C][N] fp32 -> out [b][N][C] bf16  (LDS tiled)
// ---------------------------------------------------------------------------
__global__ __launch_bounds__(256) void transpose_cvt(
    const float* __restrict__ in, __bf16* __restrict__ out, int C, int N)
{
    __shared__ float tile[32][33];
    int b  = blockIdx.z;
    int n0 = blockIdx.x << 5;
    int c0 = blockIdx.y << 5;
    int tx = threadIdx.x & 31, ty = threadIdx.x >> 5;

    const float* ip = in + ((size_t)b * C + c0) * N + n0;
#pragma unroll
    for (int r = 0; r < 4; ++r) {
        int cl = ty + 8 * r;
        tile[cl][tx] = ip[(size_t)cl * N + tx];
    }
    __syncthreads();
    __bf16* op = out + ((size_t)b * N + n0) * C + c0;
#pragma unroll
    for (int r = 0; r < 4; ++r) {
        int nl = ty + 8 * r;
        op[(size_t)nl * C + tx] = (__bf16)tile[tx][nl];
    }
}

// ---------------------------------------------------------------------------
// bf16 WMMA GEMM:  C[b] = A (MxK, packed bf16) * B[b] (Bt: [N][K] bf16)
// Block = 8 waves = 128M x 128N; wave = 16M x 128N (8 accumulators).
// B panel (128n x 32k bf16 = 8KB) staged with GLOBAL_LOAD_ASYNC_TO_LDS_B128
// (pure copy, no VGPR round-trip), double-buffered, s_wait_asynccnt fenced.
// Each lane's B fragment = two 8-bf16 runs in LDS (k 8h+0..7, 16+8h+0..7) = 2x ds_load_b128.
// Optional fused epilogue: out = xres + C*inv + (beta - mean*inv)  (BN+residual)
// ---------------------------------------------------------------------------
__global__ __launch_bounds__(256) void gemm_bf16_wmma(
    const __bf16* __restrict__ Apack, const __bf16* __restrict__ Bt,
    float* __restrict__ C, int M, int N, int K,
    long long sB, long long sC,
    const float* __restrict__ xres, const float* __restrict__ gamma,
    const float* __restrict__ beta, const float* __restrict__ mean,
    const float* __restrict__ var)
{
    // [n][k] bf16 tile: 128 rows x 32 k = 64B per row, 8KB per buffer
    __shared__ __bf16 ldsB[2][128 * 32];

    const int tid  = threadIdx.x;
    const int lane = tid & 31;                 // wave32
    const int wv   = tid >> 5;                 // 8 waves
    const int half = lane >> 4;
    const int l    = lane & 15;

    const int n0 = blockIdx.x << 7;            // 128-wide N tile
    const int m0 = (blockIdx.y << 7) + (wv << 4);
    const int mt = m0 >> 4;                    // A-pack m-tile index
    const int b  = blockIdx.z;
    const int ksteps = K >> 5;

    const __bf16* Bb = Bt + (size_t)b * (size_t)sB;

    v8f acc[8] = {};

    // Async-stage B tile: rows n0..n0+127, k-cols kb..kb+31 (64B per row)
    auto stage = [&](int s, int buf) {
        int kb = s << 5;
        unsigned base = (unsigned)(size_t)(&ldsB[buf][0]);
#pragma unroll
        for (int u = 0; u < 2; ++u) {
            int chunk = tid * 2 + u;           // 512 x 16B chunks
            int nl  = chunk >> 2;
            int seg = chunk & 3;
            const __bf16* g = Bb + (size_t)(n0 + nl) * K + kb + seg * 8;
            async_copy_b128(base + (unsigned)(nl * 64 + seg * 16), (const void*)g);
        }
    };

    stage(0, 0);
    wait_async0();
    __syncthreads();

    for (int s = 0; s < ksteps; ++s) {
        int cur = s & 1;
        if (s + 1 < ksteps) stage(s + 1, cur ^ 1);

        // A fragment: one contiguous 32B load from packed weights
        v16bf af = *reinterpret_cast<const v16bf*>(
            Apack + ((size_t)(mt * ksteps + s) * 32 + lane) * 16);

#pragma unroll
        for (int t = 0; t < 8; ++t) {
            // lane column n = t*16 + l; rows half*8..+7 then 16+half*8..+7
            const __bf16* bp = &ldsB[cur][((t << 4) + l) * 32 + (half << 3)];
            typedef __attribute__((ext_vector_type(8))) __bf16 v8bf_t;
            v8bf_t blo = *reinterpret_cast<const v8bf_t*>(bp);
            v8bf_t bhi = *reinterpret_cast<const v8bf_t*>(bp + 16);
            v16bf bf = __builtin_shufflevector(blo, bhi, 0,1,2,3,4,5,6,7,8,9,10,11,12,13,14,15);
            acc[t] = __builtin_amdgcn_wmma_f32_16x16x32_bf16(
                false, af, false, bf, (short)0, acc[t], false, false);
        }
        if (s + 1 < ksteps) wait_async0();
        __syncthreads();
    }

    // C/D layout: VGPR r, lanes 0-15 -> M = m0+r, lanes 16-31 -> M = m0+8+r
    float* Cb = C + (size_t)b * (size_t)sC;
#pragma unroll
    for (int t = 0; t < 8; ++t) {
        int n = n0 + (t << 4) + l;
#pragma unroll
        for (int r = 0; r < 8; ++r) {
            int m = m0 + r + (half << 3);
            size_t idx = (size_t)m * N + n;
            float v = acc[t][r];
            if (xres) {
                float inv = gamma[m] * rsqrtf(var[m] + 1e-6f);
                v = xres[(size_t)b * (size_t)sC + idx] + v * inv + (beta[m] - mean[m] * inv);
            }
            Cb[idx] = v;
        }
    }
}

// ---------------------------------------------------------------------------
// Depthwise 5x5 conv, pad 2, on qkv (= ms channels 0..767)
// ---------------------------------------------------------------------------
__global__ __launch_bounds__(256) void dwconv5x5(
    const float* __restrict__ ms, const float* __restrict__ wdw, float* __restrict__ out)
{
    int t = blockIdx.x * 256 + threadIdx.x;     // over BATCH*TD3*HWPX
    int n = t & (HWPX - 1);
    int bc = t >> 12;
    int c = bc % TD3;
    int b = bc / TD3;
    int y = n >> 6, x = n & 63;

    const float* ip = ms + ((size_t)b * MSCH + c) * HWPX;
    const float* wp = wdw + c * 25;
    float s = 0.f;
#pragma unroll
    for (int dy = 0; dy < 5; ++dy) {
        int yy = y + dy - 2;
        if (yy < 0 || yy > 63) continue;
#pragma unroll
        for (int dx = 0; dx < 5; ++dx) {
            int xx = x + dx - 2;
            if (xx < 0 || xx > 63) continue;
            s = fmaf(ip[yy * 64 + xx], wp[dy * 5 + dx], s);
        }
    }
    out[((size_t)b * TD3 + c) * HWPX + n] = s;
}

// ---------------------------------------------------------------------------
// Grouped pointwise agg: ms[b, 768+g*8+o, n] = sum_i w_pw[g,o,i]*dw[b,g*8+i,n]
// ---------------------------------------------------------------------------
__global__ __launch_bounds__(256) void agg_pw(
    const float* __restrict__ dw, const float* __restrict__ wpw, float* __restrict__ ms)
{
    int t = blockIdx.x * 256 + threadIdx.x;     // over BATCH*96*HWPX
    int n = t & (HWPX - 1);
    int bg = t >> 12;
    int g = bg % 96;
    int b = bg / 96;

    const float* dp = dw + ((size_t)b * TD3 + g * 8) * HWPX + n;
    float in8[8];
#pragma unroll
    for (int i = 0; i < 8; ++i) in8[i] = dp[(size_t)i * HWPX];

    const float* wp = wpw + g * 64;
    float* op = ms + ((size_t)b * MSCH + TD3 + g * 8) * HWPX + n;
#pragma unroll
    for (int o = 0; o < 8; ++o) {
        float s = 0.f;
#pragma unroll
        for (int i = 0; i < 8; ++i) s = fmaf(wp[o * 8 + i], in8[i], s);
        op[(size_t)o * HWPX] = s;
    }
}

// ---------------------------------------------------------------------------
// kv[d,e] = sum_n relu(k[n,d]) * v[n,e]   (v[:,8] == 1)
// ---------------------------------------------------------------------------
__global__ __launch_bounds__(96) void kv_reduce(
    const float* __restrict__ ms, float* __restrict__ kv)
{
    int bh = blockIdx.x;                        // 0..511
    int b = bh >> 6, h = bh & 63;
    int t = threadIdx.x;
    if (t >= 72) return;
    int d = t / 9, e = t - d * 9;

    const float* kp = ms + ((size_t)b * MSCH + 24 * h + 8 + d) * HWPX;
    float s = 0.f;
    if (e < 8) {
        const float* vp = ms + ((size_t)b * MSCH + 24 * h + 16 + e) * HWPX;
#pragma unroll 8
        for (int n = 0; n < HWPX; ++n)
            s = fmaf(fmaxf(kp[n], 0.f), vp[n], s);
    } else {
#pragma unroll 8
        for (int n = 0; n < HWPX; ++n)
            s += fmaxf(kp[n], 0.f);
    }
    kv[(size_t)bh * 72 + t] = s;
}

// ---------------------------------------------------------------------------
// out[n,e] = (sum_d relu(q[n,d]) * kv[d,e]) ; normalized by column 8.
// Writes attT[b][n][c] bf16 (c = h*8+e) as one 16B store -> proj GEMM B layout.
// ---------------------------------------------------------------------------
__global__ __launch_bounds__(256) void att_apply(
    const float* __restrict__ ms, const float* __restrict__ kv, __bf16* __restrict__ attT)
{
    __shared__ float skv[72];
    int bh = blockIdx.x >> 4;
    int chunk = blockIdx.x & 15;
    int b = bh >> 6, h = bh & 63;

    if (threadIdx.x < 72) skv[threadIdx.x] = kv[(size_t)bh * 72 + threadIdx.x];
    __syncthreads();

    int n = chunk * 256 + threadIdx.x;
    const float* qp = ms + ((size_t)b * MSCH + 24 * h) * HWPX + n;
    float q[8];
#pragma unroll
    for (int d = 0; d < 8; ++d) q[d] = fmaxf(qp[(size_t)d * HWPX], 0.f);

    float num[9];
#pragma unroll
    for (int e = 0; e < 9; ++e) {
        float s = 0.f;
#pragma unroll
        for (int d = 0; d < 8; ++d) s = fmaf(q[d], skv[d * 9 + e], s);
        num[e] = s;
    }
    float rden = 1.f / (num[8] + 1e-15f);

    uint4 o;
    o.x = pack_bf16x2(num[0] * rden, num[1] * rden);
    o.y = pack_bf16x2(num[2] * rden, num[3] * rden);
    o.z = pack_bf16x2(num[4] * rden, num[5] * rden);
    o.w = pack_bf16x2(num[6] * rden, num[7] * rden);
    *reinterpret_cast<uint4*>(attT + ((size_t)b * HWPX + n) * ATTCH + h * 8) = o;
}

// ---------------------------------------------------------------------------
extern "C" void kernel_launch(void* const* d_in, const int* in_sizes, int n_in,
                              void* d_out, int out_size, void* d_ws, size_t ws_size,
                              hipStream_t stream) {
    const float* x      = (const float*)d_in[0];
    const float* w_qkv  = (const float*)d_in[1];
    const float* w_dw   = (const float*)d_in[2];
    const float* w_pw   = (const float*)d_in[3];
    const float* w_proj = (const float*)d_in[4];
    const float* gamma  = (const float*)d_in[5];
    const float* beta   = (const float*)d_in[6];
    const float* mean   = (const float*)d_in[7];
    const float* var    = (const float*)d_in[8];
    float* out = (float*)d_out;

    // workspace layout
    const size_t MS_BYTES = (size_t)BATCH * MSCH * HWPX * sizeof(float); // 192 MB
    const size_t DW_BYTES = (size_t)BATCH * TD3 * HWPX * sizeof(float);  //  96 MB
    const size_t OFF_KV   = MS_BYTES + DW_BYTES;
    const size_t OFF_PQ   = OFF_KV + (1u << 20);      // packed w_qkv (393 KB)
    const size_t OFF_PP   = OFF_PQ + (1u << 20);      // packed w_proj (262 KB)
    const size_t OFF_XT   = OFF_PP + (1u << 20);      // xT bf16 (16.8 MB)

    float*  ms   = (float*)d_ws;
    float*  dw   = (float*)((char*)d_ws + MS_BYTES);
    __bf16* attT = (__bf16*)dw;                        // reuse: dw dead after agg
    float*  kv   = (float*)((char*)d_ws + OFF_KV);
    __bf16* pqkv = (__bf16*)((char*)d_ws + OFF_PQ);
    __bf16* pprj = (__bf16*)((char*)d_ws + OFF_PP);
    __bf16* xt   = (__bf16*)((char*)d_ws + OFF_XT);

    (void)in_sizes; (void)n_in; (void)out_size; (void)ws_size;

    // 0) pack weights into WMMA fragment order (bf16)
    {
        int thq = (TD3 / 16) * (CIN / 32) * 32;        // 12288
        pack_wgt<<<(thq + 255) / 256, 256, 0, stream>>>(w_qkv, pqkv, TD3, CIN);
        int thp = (CIN / 16) * (ATTCH / 32) * 32;      // 8192
        pack_wgt<<<(thp + 255) / 256, 256, 0, stream>>>(w_proj, pprj, CIN, ATTCH);
    }
    // 0b) x -> xT [b][n][c] bf16 for the async-staged GEMM B operand
    {
        dim3 grid(HWPX / 32, CIN / 32, BATCH);         // 128 x 8 x 8
        transpose_cvt<<<grid, 256, 0, stream>>>(x, xt, CIN, HWPX);
    }
    // 1) qkv = w_qkv * x  -> ms channels [0,768)
    {
        dim3 grid(HWPX / 128, TD3 / 128, BATCH);       // 32 x 6 x 8
        gemm_bf16_wmma<<<grid, 256, 0, stream>>>(
            pqkv, xt, ms, TD3, HWPX, CIN,
            (long long)HWPX * CIN, (long long)MSCH * HWPX,
            nullptr, nullptr, nullptr, nullptr, nullptr);
    }
    // 2) depthwise 5x5 on qkv -> dw
    dwconv5x5<<<(BATCH * TD3 * HWPX) / 256, 256, 0, stream>>>(ms, w_dw, dw);
    // 3) grouped pointwise -> ms channels [768,1536)
    agg_pw<<<(BATCH * 96 * HWPX) / 256, 256, 0, stream>>>(dw, w_pw, ms);
    // 4) kv reduction per (b,head)
    kv_reduce<<<BATCH * 64, 96, 0, stream>>>(ms, kv);
    // 5) attention apply -> attT bf16 [b][n][c] (overwrites dw region)
    att_apply<<<BATCH * 64 * 16, 256, 0, stream>>>(ms, kv, attT);
    // 6) proj + BN + residual -> d_out
    {
        dim3 grid(HWPX / 128, CIN / 128, BATCH);       // 32 x 2 x 8
        gemm_bf16_wmma<<<grid, 256, 0, stream>>>(
            pprj, attT, out, CIN, HWPX, ATTCH,
            (long long)HWPX * ATTCH, (long long)CIN * HWPX,
            x, gamma, beta, mean, var);
    }
}
